// Discriminator_80659485818928
// MI455X (gfx1250) — matrix operand order repaired and checked by reading, not compiled, not run on hardware
//
#include <hip/hip_runtime.h>
#include <hip/hip_bf16.h>

// ---------------------------------------------------------------------------
// HouseGAN-style discriminator for MI455X (gfx1250, wave32, WMMA).
// All 3x3 convs run as implicit-GEMM on v_wmma_f32_16x16x32_f16.
// Graph pooling (segment_sum over edges) is a dense 256x256 adjacency GEMM,
// also on WMMA. Boundary pyramid computed once (shared across all nodes).
//
// Round-4 change: the adjacency GEMM stages its B panel with the CDNA5
// async-to-LDS engine (global_load_async_to_lds_b32 + s_wait_asynccnt),
// double-buffered, with the fp32->f16 convert folded into the fragment load.
// Conv kernel unchanged (round-3 disasm showed a clean pipelined loop).
// ---------------------------------------------------------------------------

typedef _Float16 v16h __attribute__((ext_vector_type(16)));
typedef _Float16 v8h  __attribute__((ext_vector_type(8)));
typedef float    v8f  __attribute__((ext_vector_type(8)));

#define MODE_V 0   // per-node spatial tensor  [V,C,H,W]
#define MODE_B 1   // broadcast spatial tensor [C,H,W] (shared by all nodes)
#define MODE_S 2   // per-node channel scalar  [V,C]   (broadcast over H,W)

struct SrcDesc { const float* p; int ch; int mode; };

__device__ __forceinline__ void pack2(_Float16* dst, float a, float b,
                                      bool oka, bool okb) {
  union { _Float16 h[2]; unsigned u; } pk;
  pk.h[0] = oka ? (_Float16)a : (_Float16)0.0f;
  pk.h[1] = okb ? (_Float16)b : (_Float16)0.0f;
  *(unsigned*)dst = pk.u;
}
__device__ __forceinline__ void pack2u(_Float16* dst, float a, float b) {
  union { _Float16 h[2]; unsigned u; } pk;
  pk.h[0] = (_Float16)a; pk.h[1] = (_Float16)b;
  *(unsigned*)dst = pk.u;
}

// Per-(K-index) channel info, computed once per K-step per thread.
struct GKInfo {
  const float* base;  // (v,ci) plane start (spatial) or the scalar element
  int kh, kw;
  bool kvalid;
  bool scalar;
};

__device__ __forceinline__ GKInfo make_gkinfo(
    const SrcDesc& s0, const SrcDesc& s1, const SrcDesc& s2,
    int v, int gk, int K, int Hin, int Win) {
  GKInfo gi;
  int gkc = gk < K - 1 ? gk : K - 1;
  int ci = gkc / 9, r9 = gkc - ci * 9;
  gi.kh = r9 / 3; gi.kw = r9 - gi.kh * 3;
  gi.kvalid = gk < K;
  const bool u1 = ci >= s0.ch;
  const int  c01 = s0.ch + s1.ch;
  const bool u2 = ci >= c01;
  const float* sp = u2 ? s2.p : (u1 ? s1.p : s0.p);
  int lc   = u2 ? ci - c01 : (u1 ? ci - s0.ch : ci);
  int mode = u2 ? s2.mode  : (u1 ? s1.mode   : s0.mode);
  int sch  = u2 ? s2.ch    : (u1 ? s1.ch     : s0.ch);
  unsigned vv = (mode == MODE_B) ? 0u : (unsigned)v;
  unsigned plane = vv * (unsigned)sch + (unsigned)lc;
  gi.scalar = (mode == MODE_S);
  gi.base = gi.scalar ? (sp + plane) : (sp + plane * (unsigned)(Hin * Win));
  return gi;
}

__device__ __forceinline__ float gk_load(const GKInfo& gi, int oy, int ox,
                                         int stride, int Hin, int Win,
                                         bool pvalid, bool& ok) {
  int iy = oy * stride + gi.kh - 1, ix = ox * stride + gi.kw - 1;
  ok = gi.kvalid & pvalid & (iy >= 0) & (iy < Hin) & (ix >= 0) & (ix < Win);
  int iyc = iy < 0 ? 0 : (iy >= Hin ? Hin - 1 : iy);
  int ixc = ix < 0 ? 0 : (ix >= Win ? Win - 1 : ix);
  unsigned off = gi.scalar ? 0u : (unsigned)(iyc * Win + ixc);
  return gi.base[off];
}

// Fragment loads: two 16B contiguous LDS runs per operand (contiguous k order: lane half h holds K 16h..16h+15, same for A and B).
__device__ __forceinline__ v16h load_afrag(const _Float16* As /*pitch 40*/,
                                           int am, int lane) {
  const int k0 = (lane < 16) ? 0 : 16;  // lanes<16: K 0..15
  const _Float16* ap = As + am * 40;
  v8h lo = *(const v8h*)(ap + k0);
  v8h hi = *(const v8h*)(ap + 8 + k0);
  return __builtin_shufflevector(lo, hi, 0, 1, 2, 3, 4, 5, 6, 7,
                                 8, 9, 10, 11, 12, 13, 14, 15);
}
__device__ __forceinline__ v16h load_bfrag(const _Float16* Bt /*pitch 40*/,
                                           int bn, int lane) {
  const int bk = (lane < 16) ? 0 : 16;  // lanes<16: K 0..15
  const _Float16* bp = Bt + bn * 40 + bk;
  v8h lo = *(const v8h*)(bp);
  v8h hi = *(const v8h*)(bp + 8);
  return __builtin_shufflevector(lo, hi, 0, 1, 2, 3, 4, 5, 6, 7,
                                 8, 9, 10, 11, 12, 13, 14, 15);
}
// fp32 B fragment (async-staged panel), convert folded in (cvt_pk in codegen).
__device__ __forceinline__ v16h load_bfrag32(const float* Bs /*pitch 36*/,
                                             int bn, int lane) {
  const int bk = (lane < 16) ? 0 : 16;
  const float* bp = Bs + bn * 36 + bk;
  float4 f0 = *(const float4*)(bp);
  float4 f1 = *(const float4*)(bp + 4);
  float4 f2 = *(const float4*)(bp + 8);
  float4 f3 = *(const float4*)(bp + 12);
  v16h r;
  r[0]  = (_Float16)f0.x; r[1]  = (_Float16)f0.y;
  r[2]  = (_Float16)f0.z; r[3]  = (_Float16)f0.w;
  r[4]  = (_Float16)f1.x; r[5]  = (_Float16)f1.y;
  r[6]  = (_Float16)f1.z; r[7]  = (_Float16)f1.w;
  r[8]  = (_Float16)f2.x; r[9]  = (_Float16)f2.y;
  r[10] = (_Float16)f2.z; r[11] = (_Float16)f2.w;
  r[12] = (_Float16)f3.x; r[13] = (_Float16)f3.y;
  r[14] = (_Float16)f3.z; r[15] = (_Float16)f3.w;
  return r;
}

// Per-K-step staged registers (loads issued long before use).
template <int MT>
struct Stage {
  static constexpr int AP = MT;       // packed pairs of A per thread
  static constexpr int BP = 8 / MT;   // packed pairs of B per thread
  float a0[AP], a1[AP];
  bool  ao0[AP], ao1[AP];
  float b0[BP], b1[BP];
  bool  bo0[BP], bo1[BP];
};

template <int MT>
__device__ __forceinline__ void conv_load_stage(
    Stage<MT>& s, int kk, int kfix, int row0,
    const SrcDesc& s0, const SrcDesc& s1, const SrcDesc& s2,
    const float* __restrict__ w,
    int v, int mbase, int pbase,
    int K, int P, int Hin, int Win, int Wout, int stride) {
  const int gk0 = kk + kfix, gk1 = gk0 + 1;
  // weights (gm always < Cout: Cout is a multiple of the M tile in this net)
#pragma unroll
  for (int e = 0; e < Stage<MT>::AP; ++e) {
    int gm = mbase + row0 + e * 16;
    bool o0 = gk0 < K, o1 = gk1 < K;
    unsigned base = (unsigned)gm * (unsigned)K;
    s.ao0[e] = o0; s.ao1[e] = o1;
    s.a0[e] = w[base + (unsigned)(o0 ? gk0 : K - 1)];
    s.a1[e] = w[base + (unsigned)(o1 ? gk1 : K - 1)];
    if (e == 0 && kk + 32 < K)
      __builtin_prefetch(&w[base + (unsigned)gk0 + 32u], 0, 1);
  }
  // im2col: channel decode + source select hoisted (k invariant per thread)
  GKInfo g0 = make_gkinfo(s0, s1, s2, v, gk0, K, Hin, Win);
  GKInfo g1 = make_gkinfo(s0, s1, s2, v, gk1, K, Hin, Win);
#pragma unroll
  for (int e = 0; e < Stage<MT>::BP; ++e) {
    int p = pbase + row0 + e * 16;
    bool pv = p < P;
    int pc = pv ? p : P - 1;
    int oy = pc / Wout, ox = pc - oy * Wout;
    s.b0[e] = gk_load(g0, oy, ox, stride, Hin, Win, pv, s.bo0[e]);
    s.b1[e] = gk_load(g1, oy, ox, stride, Hin, Win, pv, s.bo1[e]);
  }
}

template <int MT>
__device__ __forceinline__ void conv_pack_stage(
    const Stage<MT>& s, int kfix, int row0,
    _Float16 (*As)[40], _Float16 (*Bt)[40]) {
#pragma unroll
  for (int e = 0; e < Stage<MT>::AP; ++e)
    pack2(&As[row0 + e * 16][kfix], s.a0[e], s.a1[e], s.ao0[e], s.ao1[e]);
#pragma unroll
  for (int e = 0; e < Stage<MT>::BP; ++e)
    pack2(&Bt[row0 + e * 16][kfix], s.b0[e], s.b1[e], s.bo0[e], s.bo1[e]);
}

// ---------------------------------------------------------------------------
// Fused conv3x3 (pad=1, stride 1|2) + bias (+ optional LeakyReLU) via WMMA.
// out[Cout, P] = W[Cout, Cin*9] x im2col[Cin*9, P], per image v.
// 256 threads = 8 wave32. MT=2 -> 32x64 tile, MT=1 -> 16x128 tile.
// Double-buffered LDS; one barrier per K-step.
// ---------------------------------------------------------------------------
template <int MT>
__global__ __launch_bounds__(256)
void conv3x3_wmma(SrcDesc s0, SrcDesc s1, SrcDesc s2,
                  const float* __restrict__ w, const float* __restrict__ bias,
                  float* __restrict__ out,
                  int Hin, int Win, int Hout, int Wout, int stride,
                  int Cin, int Cout, int leaky) {
  constexpr int NT = 8 / MT;
  constexpr int NCOLS = NT * 16;
  const int v = blockIdx.z;
  const int mbase = blockIdx.y * 16 * MT;
  const int pbase = blockIdx.x * NCOLS;
  const int K = Cin * 9;
  const int P = Hout * Wout;

  __shared__ alignas(16) _Float16 As[2][MT * 16][40];
  __shared__ alignas(16) _Float16 Bt[2][NCOLS][40];

  const int tid = threadIdx.x;
  const int wave = tid >> 5, lane = tid & 31;
  const int mi = wave / NT, ni = wave % NT;
  const int kfix = (tid & 15) * 2;
  const int row0 = tid >> 4;
  const int am = mi * 16 + (lane & 15);
  const int bn = ni * 16 + (lane & 15);

  v8f acc;
#pragma unroll
  for (int r = 0; r < 8; ++r) acc[r] = 0.0f;

  Stage<MT> s;
  conv_load_stage<MT>(s, 0, kfix, row0, s0, s1, s2, w, v, mbase, pbase,
                      K, P, Hin, Win, Wout, stride);
  conv_pack_stage<MT>(s, kfix, row0, As[0], Bt[0]);
  __syncthreads();

  int cur = 0;
  for (int kk = 32; kk < K; kk += 32) {
    conv_load_stage<MT>(s, kk, kfix, row0, s0, s1, s2, w, v, mbase, pbase,
                        K, P, Hin, Win, Wout, stride);   // loads in flight
    v16h a = load_afrag(&As[cur][0][0], am, lane);
    v16h b = load_bfrag(&Bt[cur][0][0], bn, lane);
    acc = __builtin_amdgcn_wmma_f32_16x16x32_f16(false, a, false, b,
                                                 (short)0, acc, false, false);
    conv_pack_stage<MT>(s, kfix, row0, As[cur ^ 1], Bt[cur ^ 1]);
    __syncthreads();
    cur ^= 1;
  }
  {
    v16h a = load_afrag(&As[cur][0][0], am, lane);
    v16h b = load_bfrag(&Bt[cur][0][0], bn, lane);
    acc = __builtin_amdgcn_wmma_f32_16x16x32_f16(false, a, false, b,
                                                 (short)0, acc, false, false);
  }

  // Branchless epilogue: Cout is always a multiple of the M tile here, so no
  // row check; one col check; 32-bit addressing with strength-reduced stride.
  const int col = pbase + bn;
  const int m0 = mbase + mi * 16 + ((lane >> 4) << 3);
  const float4 z0 = *(const float4*)(bias + m0);
  const float4 z1 = *(const float4*)(bias + m0 + 4);
  float bv[8] = {z0.x, z0.y, z0.z, z0.w, z1.x, z1.y, z1.z, z1.w};
  if (col < P) {
    float* po = out + ((unsigned)v * (unsigned)Cout + (unsigned)m0) *
                          (unsigned)P + (unsigned)col;
#pragma unroll
    for (int r = 0; r < 8; ++r) {
      float valf = acc[r] + bv[r];
      if (leaky) valf = valf > 0.0f ? valf : 0.1f * valf;
      *po = valf;
      po += P;
    }
  }
}

// ---------------------------------------------------------------------------
// Dense GEMM C[M,N] = A[M,K] x B[K,N] (fp32 in/out, f16 WMMA math).
// REQUIRES M%32==0, K%32==0, N%64==0 (true for all uses here) -> no masks.
// B panel staged with CDNA5 async-to-LDS (transposed scatter, fp32),
// double-buffered; A panel staged via registers as packed f16.
// ---------------------------------------------------------------------------
struct GA { float a0[2], a1[2]; };

__device__ __forceinline__ void gemm_loadA(GA& s, const float* __restrict__ Am,
                                           int kk, int kfix, int row0,
                                           int mbase, int K) {
  const int gk = kk + kfix;
#pragma unroll
  for (int e = 0; e < 2; ++e) {
    unsigned base = (unsigned)(mbase + row0 + e * 16) * (unsigned)K + (unsigned)gk;
    s.a0[e] = Am[base]; s.a1[e] = Am[base + 1];
  }
}
__device__ __forceinline__ void gemm_packA(const GA& s, int kfix, int row0,
                                           _Float16 (*As)[40]) {
#pragma unroll
  for (int e = 0; e < 2; ++e) pack2u(&As[row0 + e * 16][kfix], s.a0[e], s.a1[e]);
}

// Async-stage the 32x64 B tile, transposed into Bs32[n][k] (pitch 36 floats).
// Consecutive lanes read consecutive n (coalesced); each lane scatters its
// dword to the transposed LDS slot. Tracked by ASYNCcnt.
__device__ __forceinline__ void gemm_asyncB(const float* __restrict__ Bm,
                                            int kk, int tid, int nbase, int N,
                                            unsigned lds_base_bytes) {
#pragma unroll
  for (int e = 0; e < 8; ++e) {
    int idx = tid + e * 256;
    int k = idx >> 6, n = idx & 63;
    unsigned long long gaddr = (unsigned long long)(const void*)
        (Bm + (unsigned)(kk + k) * (unsigned)N + (unsigned)(nbase + n));
    unsigned laddr = lds_base_bytes + ((unsigned)n * 36u + (unsigned)k) * 4u;
    asm volatile("global_load_async_to_lds_b32 %0, %1, off"
                 :: "v"(laddr), "v"(gaddr) : "memory");
  }
}

__global__ __launch_bounds__(256)
void gemm_wmma(const float* __restrict__ Am, const float* __restrict__ Bm,
               float* __restrict__ Cm, int M, int K, int N) {
  __shared__ alignas(16) _Float16 As[2][32][40];
  __shared__ alignas(16) float Bs32[2][64][36];
  const int tid = threadIdx.x;
  const int wave = tid >> 5, lane = tid & 31;
  const int mi = wave >> 2, ni = wave & 3;
  const int mbase = blockIdx.y * 32, nbase = blockIdx.x * 64;
  const int kfix = (tid & 15) * 2;
  const int row0 = tid >> 4;
  const int am = mi * 16 + (lane & 15);
  const int bn = ni * 16 + (lane & 15);
  // LDS byte offset of the B staging buffer (generic LDS addr low 32 bits).
  const unsigned ldsB = (unsigned)(unsigned long long)(const void*)&Bs32[0][0][0];
  const unsigned bufBytes = 64u * 36u * 4u;

  v8f acc;
#pragma unroll
  for (int r = 0; r < 8; ++r) acc[r] = 0.0f;

  GA s;
  gemm_asyncB(Bm, 0, tid, nbase, N, ldsB);
  gemm_loadA(s, Am, 0, kfix, row0, mbase, K);
  gemm_packA(s, kfix, row0, As[0]);
  asm volatile("s_wait_asynccnt 0" ::: "memory");
  __syncthreads();

  int cur = 0;
  for (int kk = 32; kk < K; kk += 32) {
    gemm_asyncB(Bm, kk, tid, nbase, N, ldsB + (unsigned)(cur ^ 1) * bufBytes);
    gemm_loadA(s, Am, kk, kfix, row0, mbase, K);
    v16h a = load_afrag(&As[cur][0][0], am, lane);
    v16h b = load_bfrag32(&Bs32[cur][0][0], bn, lane);
    acc = __builtin_amdgcn_wmma_f32_16x16x32_f16(false, a, false, b,
                                                 (short)0, acc, false, false);
    gemm_packA(s, kfix, row0, As[cur ^ 1]);
    asm volatile("s_wait_asynccnt 0" ::: "memory");
    __syncthreads();
    cur ^= 1;
  }
  {
    v16h a = load_afrag(&As[cur][0][0], am, lane);
    v16h b = load_bfrag32(&Bs32[cur][0][0], bn, lane);
    acc = __builtin_amdgcn_wmma_f32_16x16x32_f16(false, a, false, b,
                                                 (short)0, acc, false, false);
  }
  float* po = Cm + (unsigned)(mbase + mi * 16 + ((lane >> 4) << 3)) *
                       (unsigned)N + (unsigned)(nbase + bn);
#pragma unroll
  for (int r = 0; r < 8; ++r) { *po = acc[r]; po += N; }
}

// ---------------------------------------------------------------------------
// GroupNorm(1,C) per sample + per-channel affine + LeakyReLU(0.1). In-place.
// Folded per-channel FMA; channel-major apply loop when HW >= 256.
// ---------------------------------------------------------------------------
__global__ __launch_bounds__(256)
void gn_leaky(const float* __restrict__ in, float* __restrict__ out,
              const float* __restrict__ g, const float* __restrict__ be,
              int C, int HW) {
  const int v = blockIdx.x;
  const unsigned n = (unsigned)C * (unsigned)HW;
  const float* ip = in + (size_t)v * n;
  float* op = out + (size_t)v * n;
  __shared__ float s1[256], s2[256];
  float a = 0.0f, b = 0.0f;
  for (unsigned i = threadIdx.x; i < n; i += 256) {
    float x = ip[i];
    a += x; b += x * x;
  }
  s1[threadIdx.x] = a; s2[threadIdx.x] = b;
  __syncthreads();
  for (int st = 128; st > 0; st >>= 1) {
    if (threadIdx.x < st) { s1[threadIdx.x] += s1[threadIdx.x + st];
                            s2[threadIdx.x] += s2[threadIdx.x + st]; }
    __syncthreads();
  }
  const float mu  = s1[0] / n;
  const float var = s2[0] / n - mu * mu;
  const float inv = rsqrtf(var + 1e-5f);
  if (HW >= 256) {
    for (int c = 0; c < C; ++c) {
      const float sc = g[c] * inv;
      const float of = be[c] - mu * sc;
      const float* cip = ip + (unsigned)c * HW;
      float* cop = op + (unsigned)c * HW;
      for (int i = threadIdx.x; i < HW; i += 256) {
        float x = cip[i] * sc + of;
        cop[i] = x > 0.0f ? x : 0.1f * x;
      }
    }
  } else {
    for (unsigned i = threadIdx.x; i < n; i += 256) {
      int c = i / (unsigned)HW;
      float x = (ip[i] - mu) * inv * g[c] + be[c];
      op[i] = x > 0.0f ? x : 0.1f * x;
    }
  }
}

// Adjacency build: A[dst,src] += 1/E and A[src,dst] += 1/E per edge.
__global__ void build_adj(const int* __restrict__ gw, float* __restrict__ A,
                          int E, int V) {
  int e = blockIdx.x * blockDim.x + threadIdx.x;
  if (e < E) {
    int s = gw[e * 3 + 0], d = gw[e * 3 + 2];
    float inv = 1.0f / (float)E;
    atomicAdd(&A[d * V + s], inv);
    atomicAdd(&A[s * V + d], inv);
  }
}

// y[v,o] = [given_y | topo] @ l1_w.T + l1_b   (256x40 @ 40x32)
__global__ void linear40(const float* __restrict__ gy, const float* __restrict__ tv,
                         const float* __restrict__ W, const float* __restrict__ b,
                         float* __restrict__ y) {
  int i = blockIdx.x * blockDim.x + threadIdx.x;
  if (i < 256 * 32) {
    int v = i >> 5, o = i & 31;
    float s = b[o];
#pragma unroll
    for (int j = 0; j < 8; ++j)  s += gy[v * 8 + j]  * W[o * 40 + j];
#pragma unroll
    for (int j = 0; j < 32; ++j) s += tv[v * 32 + j] * W[o * 40 + 8 + j];
    y[i] = s;
  }
}

// out[v,c,p] = c<16 ? h[v,c,p] : img8[c-16,p]   (h:[256,16,8,8], img8:[16,8,8])
__global__ void concat_img8(const float* __restrict__ h, const float* __restrict__ img8,
                            float* __restrict__ out) {
  int i = blockIdx.x * blockDim.x + threadIdx.x;
  if (i < 256 * 32 * 64) {
    int v = i / 2048, r = i - v * 2048;
    int c = r >> 6, p = r & 63;
    out[i] = (c < 16) ? h[v * 1024 + c * 64 + p] : img8[(c - 16) * 64 + p];
  }
}

// segment_sum of h[V, CHW] into out[B, CHW] by nd_to_sample (out pre-zeroed)
__global__ void pool_nodes(const float* __restrict__ h, const int* __restrict__ nd,
                           float* __restrict__ out, int V, int CHW) {
  int i = blockIdx.x * blockDim.x + threadIdx.x;
  if (i < V * CHW) {
    int v = i / CHW, r = i - v * CHW;
    atomicAdd(&out[nd[v] * CHW + r], h[i]);
  }
}

// out[b] = dot(xg[b],wg)+bg + dot(xl[b],wl)+bl
__global__ void final_head(const float* __restrict__ xg, const float* __restrict__ xl,
                           const float* __restrict__ wg, const float* __restrict__ bg,
                           const float* __restrict__ wl, const float* __restrict__ bl,
                           float* __restrict__ out) {
  int b = blockIdx.x;
  __shared__ float red[128];
  int t = threadIdx.x;
  red[t] = xg[b * 128 + t] * wg[t] + xl[b * 128 + t] * wl[t];
  __syncthreads();
  for (int st = 64; st > 0; st >>= 1) {
    if (t < st) red[t] += red[t + st];
    __syncthreads();
  }
  if (t == 0) out[b] = red[0] + bg[0] + bl[0];
}

// ---------------------------------------------------------------------------
// Host orchestration
// ---------------------------------------------------------------------------
struct BP { const float *w, *b, *g, *be; };

extern "C" void kernel_launch(void* const* d_in, const int* in_sizes, int n_in,
                              void* d_out, int out_size, void* d_ws, size_t ws_size,
                              hipStream_t stream) {
  const int V = 256, RES = 64, E = 1024, NB = 8;
  (void)in_sizes; (void)n_in; (void)out_size; (void)ws_size;

  // ---- inputs (setup_inputs() dict insertion order, params flattened
  //      recursively in insertion order; per-conv block order: w,b[,g,be]) ----
  const float* x_in   = (const float*)d_in[0];
  const float* giveny = (const float*)d_in[1];
  const float* topo   = (const float*)d_in[2];
  const float* gb     = (const float*)d_in[3];
  int pi = 4;
  auto f = [&](int i) { return (const float*)d_in[i]; };
  const float* l1_w = f(pi++); const float* l1_b = f(pi++);
  auto take = [&](bool gn) {
    BP p; p.w = f(pi++); p.b = f(pi++);
    if (gn) { p.g = f(pi++); p.be = f(pi++); } else { p.g = nullptr; p.be = nullptr; }
    return p;
  };
  BP enc0 = take(false), enc1 = take(true), enc2 = take(true), enc3 = take(true);
  BP cd[3];         for (int i = 0; i < 3; ++i) cd[i] = take(true);
  BP es[2];         for (int i = 0; i < 2; ++i) es[i] = take(true);
  BP ed[3][2];      for (int i = 0; i < 3; ++i) for (int j = 0; j < 2; ++j) ed[i][j] = take(true);
  BP cmpB[3][3];    for (int i = 0; i < 3; ++i) for (int j = 0; j < 3; ++j) cmpB[i][j] = take(true);
  BP gd0 = take(true), gd1 = take(true), gd2 = take(false);
  BP ld0 = take(true), ld1 = take(true), ld2 = take(false);
  const float* fcg_w = f(pi++); const float* fcg_b = f(pi++);
  const float* fcl_w = f(pi++); const float* fcl_b = f(pi++);
  const int* gw = (const int*)d_in[pi++];
  const int* nd = (const int*)d_in[pi++];

  // ---- workspace layout (floats) ----
  float* W0 = (float*)d_ws;
  size_t off = 0;
  auto alloc = [&](size_t nf) { float* p = W0 + off; off += (nf + 63) & ~(size_t)63; return p; };
  float* C32   = alloc((size_t)32 * V * RES * RES);   // 32-ch scratch (largest at 64x64)
  float* X1    = alloc((size_t)16 * V * RES * RES);
  float* X2    = alloc((size_t)16 * V * RES * RES);
  float* POOLb = alloc((size_t)16 * V * RES * RES);
  float* ADJ   = alloc((size_t)V * V);
  float* Y32   = alloc((size_t)V * 32);
  float* IMG64 = alloc((size_t)16 * 64 * 64);
  float* IMG32 = alloc((size_t)16 * 32 * 32);
  float* IMG16 = alloc((size_t)16 * 16 * 16);
  float* IMG8  = alloc((size_t)16 * 8 * 8);
  float* TMPI  = alloc((size_t)16 * 64 * 64);
  float* HCAT  = alloc((size_t)V * 32 * 64);
  float* XG0   = alloc((size_t)NB * 32 * 64);
  float* XG1   = alloc((size_t)NB * 64 * 16);
  float* XG2   = alloc((size_t)NB * 128 * 4);
  float* XG3   = alloc((size_t)NB * 128);
  float* XL1   = alloc((size_t)V * 64 * 16);
  float* XL2   = alloc((size_t)V * 128 * 4);
  float* XL3   = alloc((size_t)V * 128);
  float* XLP   = alloc((size_t)NB * 128);

  SrcDesc SN = {nullptr, 0, MODE_V};
  auto conv = [&](SrcDesc s0, SrcDesc s1, SrcDesc s2, const BP& p, float* outp,
                  int Vb, int Hin, int Win, int stride, int Cout, int leaky) {
    int Cin = s0.ch + s1.ch + s2.ch;
    int Hout = (stride == 2) ? Hin / 2 : Hin;
    int Wout = (stride == 2) ? Win / 2 : Win;
    int P = Hout * Wout;
    if (Cout > 16) {
      dim3 grid((P + 63) / 64, Cout / 32, Vb);
      conv3x3_wmma<2><<<grid, 256, 0, stream>>>(s0, s1, s2, p.w, p.b, outp,
                                                Hin, Win, Hout, Wout, stride,
                                                Cin, Cout, leaky);
    } else {
      dim3 grid((P + 127) / 128, Cout / 16, Vb);
      conv3x3_wmma<1><<<grid, 256, 0, stream>>>(s0, s1, s2, p.w, p.b, outp,
                                                Hin, Win, Hout, Wout, stride,
                                                Cin, Cout, leaky);
    }
  };
  auto gn = [&](float* buf, const BP& p, int Vb, int C, int HW) {
    gn_leaky<<<Vb, 256, 0, stream>>>(buf, buf, p.g, p.be, C, HW);
  };
  auto gemm = [&](const float* A, const float* Bm, float* Cm, int M, int K, int N) {
    dim3 grid(N / 64, M / 32, 1);
    gemm_wmma<<<grid, 256, 0, stream>>>(A, Bm, Cm, M, K, N);
  };

  // ---- graph adjacency + conditioning vector ----
  hipMemsetAsync(ADJ, 0, (size_t)V * V * sizeof(float), stream);
  build_adj<<<(E + 255) / 256, 256, 0, stream>>>(gw, ADJ, E, V);
  linear40<<<(V * 32 + 255) / 256, 256, 0, stream>>>(giveny, topo, l1_w, l1_b, Y32);

  // ---- encoder (virtual concat of x[1ch] and broadcast conditioning[32ch]) ----
  conv({x_in, 1, MODE_V}, {Y32, 32, MODE_S}, SN, enc0, X1, V, 64, 64, 1, 16, 1);
  conv({X1, 16, MODE_V}, SN, SN, enc1, X2, V, 64, 64, 1, 16, 0); gn(X2, enc1, V, 16, 4096);
  conv({X2, 16, MODE_V}, SN, SN, enc2, X1, V, 64, 64, 1, 16, 0); gn(X1, enc2, V, 16, 4096);
  conv({X1, 16, MODE_V}, SN, SN, enc3, X2, V, 64, 64, 1, 16, 0); gn(X2, enc3, V, 16, 4096);
  // h64 = X2

  // ---- boundary pyramid: identical for every node -> compute ONCE (V=1) ----
  conv({gb, 1, MODE_V},     SN, SN, es[0],    TMPI,  1, 64, 64, 1, 16, 0); gn(TMPI,  es[0],    1, 16, 4096);
  conv({TMPI, 16, MODE_V},  SN, SN, es[1],    IMG64, 1, 64, 64, 1, 16, 0); gn(IMG64, es[1],    1, 16, 4096);
  conv({IMG64, 16, MODE_V}, SN, SN, ed[0][0], TMPI,  1, 64, 64, 2, 16, 0); gn(TMPI,  ed[0][0], 1, 16, 1024);
  conv({TMPI, 16, MODE_V},  SN, SN, ed[0][1], IMG32, 1, 32, 32, 1, 16, 0); gn(IMG32, ed[0][1], 1, 16, 1024);
  conv({IMG32, 16, MODE_V}, SN, SN, ed[1][0], TMPI,  1, 32, 32, 2, 16, 0); gn(TMPI,  ed[1][0], 1, 16, 256);
  conv({TMPI, 16, MODE_V},  SN, SN, ed[1][1], IMG16, 1, 16, 16, 1, 16, 0); gn(IMG16, ed[1][1], 1, 16, 256);
  conv({IMG16, 16, MODE_V}, SN, SN, ed[2][0], TMPI,  1, 16, 16, 2, 16, 0); gn(TMPI,  ed[2][0], 1, 16, 64);
  conv({TMPI, 16, MODE_V},  SN, SN, ed[2][1], IMG8,  1, 8, 8, 1, 16, 0);   gn(IMG8,  ed[2][1], 1, 16, 64);

  // ---- CMP level 64 (h=X2) ----
  gemm(ADJ, X2, POOLb, V, V, 16 * 4096);
  conv({X2, 16, MODE_V}, {POOLb, 16, MODE_V}, {IMG64, 16, MODE_B}, cmpB[0][0], C32, V, 64, 64, 1, 32, 0);
  gn(C32, cmpB[0][0], V, 32, 4096);
  conv({C32, 32, MODE_V}, SN, SN, cmpB[0][1], X1, V, 64, 64, 1, 16, 0); gn(X1, cmpB[0][1], V, 16, 4096);
  conv({X1, 16, MODE_V},  SN, SN, cmpB[0][2], X2, V, 64, 64, 1, 16, 0); gn(X2, cmpB[0][2], V, 16, 4096);
  conv({X2, 16, MODE_V},  SN, SN, cd[0],      X1, V, 64, 64, 2, 16, 0); gn(X1, cd[0],      V, 16, 1024);
  // h32 = X1

  // ---- CMP level 32 ----
  gemm(ADJ, X1, POOLb, V, V, 16 * 1024);
  conv({X1, 16, MODE_V}, {POOLb, 16, MODE_V}, {IMG32, 16, MODE_B}, cmpB[1][0], C32, V, 32, 32, 1, 32, 0);
  gn(C32, cmpB[1][0], V, 32, 1024);
  conv({C32, 32, MODE_V}, SN, SN, cmpB[1][1], X2, V, 32, 32, 1, 16, 0); gn(X2, cmpB[1][1], V, 16, 1024);
  conv({X2, 16, MODE_V},  SN, SN, cmpB[1][2], X1, V, 32, 32, 1, 16, 0); gn(X1, cmpB[1][2], V, 16, 1024);
  conv({X1, 16, MODE_V},  SN, SN, cd[1],      X2, V, 32, 32, 2, 16, 0); gn(X2, cd[1],      V, 16, 256);
  // h16 = X2

  // ---- CMP level 16 ----
  gemm(ADJ, X2, POOLb, V, V, 16 * 256);
  conv({X2, 16, MODE_V}, {POOLb, 16, MODE_V}, {IMG16, 16, MODE_B}, cmpB[2][0], C32, V, 16, 16, 1, 32, 0);
  gn(C32, cmpB[2][0], V, 32, 256);
  conv({C32, 32, MODE_V}, SN, SN, cmpB[2][1], X1, V, 16, 16, 1, 16, 0); gn(X1, cmpB[2][1], V, 16, 256);
  conv({X1, 16, MODE_V},  SN, SN, cmpB[2][2], X2, V, 16, 16, 1, 16, 0); gn(X2, cmpB[2][2], V, 16, 256);
  conv({X2, 16, MODE_V},  SN, SN, cd[2],      X1, V, 16, 16, 2, 16, 0); gn(X1, cd[2],      V, 16, 64);
  // h8 = X1 : [256,16,8,8]

  // ---- concat with img8 -> [256,32,8,8] ----
  concat_img8<<<(V * 32 * 64 + 255) / 256, 256, 0, stream>>>(X1, IMG8, HCAT);

  // ---- global head: pool nodes -> samples, then decode ----
  hipMemsetAsync(XG0, 0, (size_t)NB * 2048 * sizeof(float), stream);
  pool_nodes<<<(V * 2048 + 255) / 256, 256, 0, stream>>>(HCAT, nd, XG0, V, 2048);
  conv({XG0, 32, MODE_V},  SN, SN, gd0, XG1, NB, 8, 8, 2, 64, 0);  gn(XG1, gd0, NB, 64, 16);
  conv({XG1, 64, MODE_V},  SN, SN, gd1, XG2, NB, 4, 4, 2, 128, 0); gn(XG2, gd1, NB, 128, 4);
  conv({XG2, 128, MODE_V}, SN, SN, gd2, XG3, NB, 2, 2, 2, 128, 1);

  // ---- local head: decode per node, then pool ----
  conv({HCAT, 32, MODE_V}, SN, SN, ld0, XL1, V, 8, 8, 2, 64, 0);  gn(XL1, ld0, V, 64, 16);
  conv({XL1, 64, MODE_V},  SN, SN, ld1, XL2, V, 4, 4, 2, 128, 0); gn(XL2, ld1, V, 128, 4);
  conv({XL2, 128, MODE_V}, SN, SN, ld2, XL3, V, 2, 2, 2, 128, 1);
  hipMemsetAsync(XLP, 0, (size_t)NB * 128 * sizeof(float), stream);
  pool_nodes<<<(V * 128 + 255) / 256, 256, 0, stream>>>(XL3, nd, XLP, V, 128);

  // ---- final: vg + vl -> d_out[8] ----
  final_head<<<NB, 128, 0, stream>>>(XG3, XLP, fcg_w, fcg_b, fcl_w, fcl_b, (float*)d_out);
}
